// SwinBlock_73306501808905
// MI455X (gfx1250) — matrix operand order repaired and checked by reading, not compiled, not run on hardware
//
#include <hip/hip_runtime.h>
#include <hip/hip_bf16.h>

typedef __attribute__((ext_vector_type(16))) _Float16 v16h;
typedef __attribute__((ext_vector_type(8)))  _Float16 v8h;
typedef __attribute__((ext_vector_type(8)))  float    v8f;

#define NWIN  64      // tokens per 8x8 window
#define DIMC  192
#define NH_   6
#define HD_   32
#define MLPH  384
#define NWIMG 1024    // windows per image (32*32)

union V16U { v16h v; v8h h[2]; };

// ---------------- CDNA5 async global->LDS helpers (ASYNCcnt path) ----------------
__device__ __forceinline__ unsigned int lds_addr_of(const void* p) {
  // flat addresses in the LDS aperture carry the LDS byte offset in the low 32 bits
  return (unsigned int)(size_t)p;
}
__device__ __forceinline__ void async_g2l_b128(unsigned int lds_addr, const void* gaddr) {
  asm volatile("global_load_async_to_lds_b128 %0, %1, off"
               :: "v"(lds_addr), "v"((unsigned long long)(size_t)gaddr)
               : "memory");
}
__device__ __forceinline__ void wait_async0() {
  asm volatile("s_wait_asynccnt 0" ::: "memory");
}

// A fragment (16x32 f16, M x K), from row-major [M][ld] f16 in LDS.
// layout: lanes 0-15 -> M, element e -> K = half*16 + e (contiguous, matches B)
__device__ __forceinline__ v16h ldsA(const _Float16* base, int ld, int lane, int m0, int k0) {
  const _Float16* p = base + (size_t)(m0 + (lane & 15)) * ld + k0 + (lane >> 4) * 16;
  V16U r;
  r.h[0] = *(const v8h*)(p);
  r.h[1] = *(const v8h*)(p + 8);
  return r.v;
}

// B fragment (32x16 f16, K x N) where data is stored TRANSPOSED row-major as [N][ld]:
// element e -> K = k0 + (lane>>4)*16 + e (contiguous), N = n0 + (lane&15)
__device__ __forceinline__ v16h ldsBT(const _Float16* base, int ld, int lane, int n0, int k0) {
  const _Float16* p = base + (size_t)(n0 + (lane & 15)) * ld + k0 + (lane >> 4) * 16;
  V16U r;
  r.h[0] = *(const v8h*)(p);
  r.h[1] = *(const v8h*)(p + 8);
  return r.v;
}

// B fragment from pre-swizzled weight buffer: frag f = kt*(N/16)+nt,
// lane chunk = 16 contiguous halves at (f*32+lane)*16
__device__ __forceinline__ v16h ldWfrag(const _Float16* __restrict__ w, int frag, int lane) {
  return *(const v16h*)(w + ((size_t)frag * 32 + lane) * 16);
}

// ---------------- weight pre-swizzle: f32 [K][N] -> f16 WMMA B-fragment layout ----------------
__global__ void swizzle_w(const float* __restrict__ src, _Float16* __restrict__ dst,
                          int K, int N, int nfrag) {
  int gid  = blockIdx.x * blockDim.x + threadIdx.x;
  int f    = gid >> 5;
  int lane = gid & 31;
  if (f >= nfrag) return;
  int ntiles = N >> 4;
  int kt = f / ntiles, nt = f - kt * ntiles;
  int kbase = kt * 32 + (lane >> 4) * 16;
  int n = nt * 16 + (lane & 15);
  V16U r;
#pragma unroll
  for (int e = 0; e < 16; ++e)
    r.v[e] = (_Float16)src[(size_t)(kbase + e) * N + n];
  *(v16h*)(dst + ((size_t)f * 32 + lane) * 16) = r.v;
}

// ---------------- kernel 1: LN1 + shift-window gather + QKV + attention + proj + residual ----------------
struct SmemAttn {
  _Float16 hwin[NWIN * DIMC];                                         // 24KB  LN'd window tokens
  union { _Float16 qk[NWIN * 2 * DIMC]; float f32buf[NWIN * DIMC]; }; // 48KB x-stage / q|k / proj-out
  _Float16 vT[DIMC * NWIN];                                           // 24KB  V transposed: [c][m]
  float    S[NWIN * NWIN];                                            // 16KB  logits (one head)
  _Float16 P[NWIN * NWIN];                                            // 8KB   softmax probs
  _Float16 obuf[NWIN * DIMC];                                         // 24KB  attention output
};

__global__ __launch_bounds__(256, 1)
void swin_attn_kernel(const float* __restrict__ x,
                      const float* __restrict__ n1w, const float* __restrict__ n1b,
                      const _Float16* __restrict__ wqkv, const float* __restrict__ bqkv,
                      const _Float16* __restrict__ wproj, const float* __restrict__ bproj,
                      const float* __restrict__ mask,
                      float* __restrict__ h2out) {
  __shared__ SmemAttn sm;
  const int tid  = threadIdx.x;
  const int lane = tid & 31, wv = tid >> 5;
  const int win  = blockIdx.x;
  const int b    = win >> 10, w = win & (NWIMG - 1);
  const int wy   = w >> 5, wx = w & 31;

  const int rowT = tid >> 2, partT = tid & 3;   // 4 threads per token row
  const int hs = wy * 8 + (rowT >> 3), wsft = wx * 8 + (rowT & 7);
  const int oh = (hs + 4) & 255, ow = (wsft + 4) & 255;
  const size_t gRowBase = ((size_t)b * 65536 + (size_t)oh * 256 + ow) * DIMC + partT * 48;

  // ---- phase 0: async-stage shifted window rows (f32) into LDS (ASYNCcnt)
  {
    const float* src = x + gRowBase;
    float* dstl = sm.f32buf + rowT * DIMC + partT * 48;
#pragma unroll
    for (int i = 0; i < 12; ++i)
      async_g2l_b128(lds_addr_of(dstl + i * 4), src + i * 4);
    wait_async0();
  }
  __syncthreads();

  // ---- phase 1: LayerNorm -> hwin (f16)
  {
    const float* src = sm.f32buf + rowT * DIMC + partT * 48;
    float buf[48];
    float s = 0.f, s2 = 0.f;
#pragma unroll
    for (int i = 0; i < 48; ++i) { float v = src[i]; buf[i] = v; s += v; s2 += v * v; }
    s  += __shfl_xor(s, 1, 4);  s  += __shfl_xor(s, 2, 4);
    s2 += __shfl_xor(s2, 1, 4); s2 += __shfl_xor(s2, 2, 4);
    const float mean = s * (1.f / DIMC);
    const float var  = s2 * (1.f / DIMC) - mean * mean;
    const float rstd = __frsqrt_rn(var + 1e-5f);
    _Float16* dst = sm.hwin + rowT * DIMC + partT * 48;
#pragma unroll
    for (int i = 0; i < 48; ++i) {
      const int c = partT * 48 + i;
      dst[i] = (_Float16)((buf[i] - mean) * rstd * n1w[c] + n1b[c]);
    }
  }
  __syncthreads();

  // ---- phase 2: QKV GEMM [64x192]@[192x576], K=192 (6 wmma steps), 144 tiles / 8 waves
  const float qscale = 0.17677669529663687f; // HD^-0.5
  for (int t = wv; t < 4 * 36; t += 8) {
    const int mt = t / 36, nt = t - mt * 36;
    v8f acc = {};
#pragma unroll
    for (int kt = 0; kt < 6; ++kt) {
      if (kt + 1 < 6)
        __builtin_prefetch(wqkv + (((size_t)(kt + 1) * 36 + nt) * 32 + lane) * 16);
      v16h a  = ldsA(sm.hwin, DIMC, lane, mt * 16, kt * 32);
      v16h bm = ldWfrag(wqkv, kt * 36 + nt, lane);
      acc = __builtin_amdgcn_wmma_f32_16x16x32_f16(false, a, false, bm, (short)0, acc, false, false);
    }
    const int col = nt * 16 + (lane & 15);
    const float bias = bqkv[col];
    const int mBase = mt * 16 + (lane >> 4) * 8;
    // nt is wave-uniform and region boundaries are tile-aligned -> scalar branch
    if (nt < 12) {                     // Q (pre-scaled)
#pragma unroll
      for (int r = 0; r < 8; ++r)
        sm.qk[(mBase + r) * (2 * DIMC) + col] = (_Float16)((acc[r] + bias) * qscale);
    } else if (nt < 24) {              // K
#pragma unroll
      for (int r = 0; r < 8; ++r)
        sm.qk[(mBase + r) * (2 * DIMC) + col] = (_Float16)(acc[r] + bias);
    } else {                           // V, transposed [c][m]
      const int c = col - 2 * DIMC;
#pragma unroll
      for (int r = 0; r < 8; ++r)
        sm.vT[(size_t)c * NWIN + (mBase + r)] = (_Float16)(acc[r] + bias);
    }
  }
  __syncthreads();

  // ---- phase 3: per-head attention
  for (int h = 0; h < NH_; ++h) {
    // S = q @ k^T (16 tiles, K=32 -> 1 wmma)
    for (int t = wv; t < 16; t += 8) {
      const int mt = t >> 2, nt = t & 3;
      v16h a  = ldsA(sm.qk, 2 * DIMC, lane, mt * 16, h * 32);
      v16h bm = ldsBT(sm.qk, 2 * DIMC, lane, nt * 16, DIMC + h * 32);
      v8f acc = {};
      acc = __builtin_amdgcn_wmma_f32_16x16x32_f16(false, a, false, bm, (short)0, acc, false, false);
      const int n = nt * 16 + (lane & 15);
#pragma unroll
      for (int r = 0; r < 8; ++r) {
        const int m = mt * 16 + (lane >> 4) * 8 + r;
        sm.S[m * NWIN + n] = acc[r];
      }
    }
    __syncthreads();

    // masked softmax over key dim (4 threads per query row)
    {
      const float* mrow = mask + ((size_t)w * NWIN + rowT) * NWIN + partT * 16;
      const float* srow = sm.S + rowT * NWIN + partT * 16;
      float vals[16]; float mx = -3.4e38f;
#pragma unroll
      for (int i = 0; i < 16; ++i) { float v = srow[i] + mrow[i]; vals[i] = v; mx = fmaxf(mx, v); }
      mx = fmaxf(mx, __shfl_xor(mx, 1, 4)); mx = fmaxf(mx, __shfl_xor(mx, 2, 4));
      float sum = 0.f;
#pragma unroll
      for (int i = 0; i < 16; ++i) { float e = __expf(vals[i] - mx); vals[i] = e; sum += e; }
      sum += __shfl_xor(sum, 1, 4); sum += __shfl_xor(sum, 2, 4);
      const float inv = 1.f / sum;
      V16U r;
#pragma unroll
      for (int i = 0; i < 16; ++i) r.v[i] = (_Float16)(vals[i] * inv);
      _Float16* prow = sm.P + rowT * NWIN + partT * 16;
      *(v8h*)(prow)     = r.h[0];
      *(v8h*)(prow + 8) = r.h[1];
    }
    __syncthreads();

    // O = P @ V (8 tiles, one per wave; K=64 -> 2 wmma)
    {
      const int mt = wv >> 1, dt = wv & 1;
      v8f acc = {};
#pragma unroll
      for (int kt = 0; kt < 2; ++kt) {
        v16h a  = ldsA(sm.P, NWIN, lane, mt * 16, kt * 32);
        v16h bm = ldsBT(sm.vT + (size_t)h * HD_ * NWIN, NWIN, lane, dt * 16, kt * 32);
        acc = __builtin_amdgcn_wmma_f32_16x16x32_f16(false, a, false, bm, (short)0, acc, false, false);
      }
      const int col = h * HD_ + dt * 16 + (lane & 15);
#pragma unroll
      for (int r = 0; r < 8; ++r) {
        const int m = mt * 16 + (lane >> 4) * 8 + r;
        sm.obuf[m * DIMC + col] = (_Float16)acc[r];
      }
    }
    __syncthreads();
  }

  // ---- phase 4: proj GEMM [64x192]@[192x192] -> f32buf (aliases qk, now dead)
  for (int t = wv; t < 48; t += 8) {
    const int mt = t / 12, nt = t - mt * 12;
    v8f acc = {};
#pragma unroll
    for (int kt = 0; kt < 6; ++kt) {
      if (kt + 1 < 6)
        __builtin_prefetch(wproj + (((size_t)(kt + 1) * 12 + nt) * 32 + lane) * 16);
      v16h a  = ldsA(sm.obuf, DIMC, lane, mt * 16, kt * 32);
      v16h bm = ldWfrag(wproj, kt * 12 + nt, lane);
      acc = __builtin_amdgcn_wmma_f32_16x16x32_f16(false, a, false, bm, (short)0, acc, false, false);
    }
    const int col = nt * 16 + (lane & 15);
    const float bias = bproj[col];
#pragma unroll
    for (int r = 0; r < 8; ++r) {
      const int m = mt * 16 + (lane >> 4) * 8 + r;
      sm.f32buf[m * DIMC + col] = acc[r] + bias;
    }
  }
  __syncthreads();

  // ---- phase 5: window-reverse + roll-back scatter + residual add -> h2 (in d_out)
  {
    const float* resid = x + gRowBase;
    float* dst = h2out + gRowBase;
    const float* o = sm.f32buf + rowT * DIMC + partT * 48;
#pragma unroll
    for (int i = 0; i < 48; ++i) dst[i] = resid[i] + o[i];
  }
}

// ---------------- kernel 2: LN2 + fc1 + GELU + fc2 + residual (in-place on d_out) ----------------
struct SmemMlp {
  float    h2[NWIN * DIMC];     // 48KB raw rows (async-staged, reused as residual)
  _Float16 a16[NWIN * DIMC];    // 24KB LN'd activations
  _Float16 mbuf[NWIN * MLPH];   // 48KB hidden activations
};

__global__ __launch_bounds__(256, 1)
void swin_mlp_kernel(float* __restrict__ io,
                     const float* __restrict__ n2w, const float* __restrict__ n2b,
                     const _Float16* __restrict__ wfc1, const float* __restrict__ bfc1,
                     const _Float16* __restrict__ wfc2, const float* __restrict__ bfc2) {
  __shared__ SmemMlp sm;
  const int tid = threadIdx.x, lane = tid & 31, wv = tid >> 5;
  const size_t row0 = (size_t)blockIdx.x * NWIN;

  // ---- phase 0: async-stage 64 contiguous rows (48KB) into LDS (ASYNCcnt)
  {
    const char* src = (const char*)(io + row0 * DIMC);
    const unsigned int dst = lds_addr_of(sm.h2);
#pragma unroll
    for (int k = 0; k < 12; ++k)
      async_g2l_b128(dst + (unsigned)(tid * 16 + k * 4096), src + tid * 16 + k * 4096);
    wait_async0();
  }
  __syncthreads();

  // ---- phase 1: LN2 from LDS -> a16 (h2 stays resident for the residual)
  {
    const int row = tid >> 2, part = tid & 3;
    const float* src = sm.h2 + row * DIMC + part * 48;
    float buf[48]; float s = 0.f, s2 = 0.f;
#pragma unroll
    for (int i = 0; i < 48; ++i) { float v = src[i]; buf[i] = v; s += v; s2 += v * v; }
    s  += __shfl_xor(s, 1, 4);  s  += __shfl_xor(s, 2, 4);
    s2 += __shfl_xor(s2, 1, 4); s2 += __shfl_xor(s2, 2, 4);
    const float mean = s * (1.f / DIMC);
    const float var  = s2 * (1.f / DIMC) - mean * mean;
    const float rstd = __frsqrt_rn(var + 1e-5f);
    _Float16* adst = sm.a16 + row * DIMC + part * 48;
#pragma unroll
    for (int i = 0; i < 48; ++i) {
      const int c = part * 48 + i;
      adst[i] = (_Float16)((buf[i] - mean) * rstd * n2w[c] + n2b[c]);
    }
  }
  __syncthreads();

  // ---- fc1 [64x192]@[192x384] + exact GELU
  for (int t = wv; t < 4 * 24; t += 8) {
    const int mt = t / 24, nt = t - mt * 24;
    v8f acc = {};
#pragma unroll
    for (int kt = 0; kt < 6; ++kt) {
      if (kt + 1 < 6)
        __builtin_prefetch(wfc1 + (((size_t)(kt + 1) * 24 + nt) * 32 + lane) * 16);
      v16h a  = ldsA(sm.a16, DIMC, lane, mt * 16, kt * 32);
      v16h bm = ldWfrag(wfc1, kt * 24 + nt, lane);
      acc = __builtin_amdgcn_wmma_f32_16x16x32_f16(false, a, false, bm, (short)0, acc, false, false);
    }
    const int col = nt * 16 + (lane & 15);
    const float bias = bfc1[col];
#pragma unroll
    for (int r = 0; r < 8; ++r) {
      const int m = mt * 16 + (lane >> 4) * 8 + r;
      const float u = acc[r] + bias;
      const float g = 0.5f * u * (1.f + erff(u * 0.70710678118654752f));
      sm.mbuf[m * MLPH + col] = (_Float16)g;
    }
  }
  __syncthreads();

  // ---- fc2 [64x384]@[384x192] + residual -> global (rows owned exclusively by this block)
  for (int t = wv; t < 48; t += 8) {
    const int mt = t / 12, nt = t - mt * 12;
    v8f acc = {};
#pragma unroll
    for (int kt = 0; kt < 12; ++kt) {
      if (kt + 1 < 12)
        __builtin_prefetch(wfc2 + (((size_t)(kt + 1) * 12 + nt) * 32 + lane) * 16);
      v16h a  = ldsA(sm.mbuf, MLPH, lane, mt * 16, kt * 32);
      v16h bm = ldWfrag(wfc2, kt * 12 + nt, lane);
      acc = __builtin_amdgcn_wmma_f32_16x16x32_f16(false, a, false, bm, (short)0, acc, false, false);
    }
    const int col = nt * 16 + (lane & 15);
    const float bias = bfc2[col];
#pragma unroll
    for (int r = 0; r < 8; ++r) {
      const int m = mt * 16 + (lane >> 4) * 8 + r;
      io[(row0 + m) * DIMC + col] = sm.h2[m * DIMC + col] + acc[r] + bias;
    }
  }
}

extern "C" void kernel_launch(void* const* d_in, const int* in_sizes, int n_in,
                              void* d_out, int out_size, void* d_ws, size_t ws_size,
                              hipStream_t stream) {
  const float* x      = (const float*)d_in[0];
  const float* n1w    = (const float*)d_in[1];
  const float* n1b    = (const float*)d_in[2];
  const float* qkv_w  = (const float*)d_in[3];
  const float* qkv_b  = (const float*)d_in[4];
  const float* proj_w = (const float*)d_in[5];
  const float* proj_b = (const float*)d_in[6];
  const float* n2w    = (const float*)d_in[7];
  const float* n2b    = (const float*)d_in[8];
  const float* fc1_w  = (const float*)d_in[9];
  const float* fc1_b  = (const float*)d_in[10];
  const float* fc2_w  = (const float*)d_in[11];
  const float* fc2_b  = (const float*)d_in[12];
  const float* mask   = (const float*)d_in[13];
  (void)in_sizes; (void)n_in; (void)out_size; (void)ws_size;

  // workspace: pre-swizzled f16 weight fragments (frag = 512 halves)
  _Float16* wsq = (_Float16*)d_ws;         // qkv: (192/32)*(576/16) = 216 frags
  _Float16* wsp = wsq + (size_t)216 * 512; // proj: 6*12 = 72 frags
  _Float16* ws1 = wsp + (size_t)72  * 512; // fc1: 6*24 = 144 frags
  _Float16* ws2 = ws1 + (size_t)144 * 512; // fc2: 12*12 = 144 frags

  swizzle_w<<<(216 * 32 + 255) / 256, 256, 0, stream>>>(qkv_w,  wsq, 192, 576, 216);
  swizzle_w<<<(72  * 32 + 255) / 256, 256, 0, stream>>>(proj_w, wsp, 192, 192, 72);
  swizzle_w<<<(144 * 32 + 255) / 256, 256, 0, stream>>>(fc1_w,  ws1, 192, 384, 144);
  swizzle_w<<<(144 * 32 + 255) / 256, 256, 0, stream>>>(fc2_w,  ws2, 384, 192, 144);

  swin_attn_kernel<<<4096, 256, 0, stream>>>(x, n1w, n1b, wsq, qkv_b, wsp, proj_b, mask,
                                             (float*)d_out);
  swin_mlp_kernel<<<4096, 256, 0, stream>>>((float*)d_out, n2w, n2b, ws1, fc1_b, ws2, fc2_b);
}
